// AVSL_Graph_38732015075736
// MI455X (gfx1250) — matrix operand order repaired and checked by reading, not compiled, not run on hardware
//
#include <hip/hip_runtime.h>
#include <hip/hip_bf16.h>
#include <stdint.h>

// ---------------------------------------------------------------------------
// CDNA5 (gfx1250) WMMA types
// ---------------------------------------------------------------------------
typedef __attribute__((ext_vector_type(16))) __bf16 v16bf;
typedef __attribute__((ext_vector_type(8)))  float  v8f;
typedef __attribute__((ext_vector_type(2)))  float  v2f;

// round-to-nearest-even f32 -> bf16 (returns 16-bit pattern in low bits)
__device__ __forceinline__ uint32_t bf16_rne(float x) {
    uint32_t u = __float_as_uint(x);
    return (u + 0x7fffu + ((u >> 16) & 1u)) >> 16;
}
// split-bf16: x ~= hi + lo with ~2^-16 relative residual
__device__ __forceinline__ void split_bf16(float x, uint32_t& hb, uint32_t& lb) {
    hb = bf16_rne(x);
    float hf = __uint_as_float(hb << 16);
    lb = bf16_rne(x - hf);
}

// ---------------------------------------------------------------------------
// Prep: split w (fp32, row-major [R][C]) into pair-packed bf16 hi/lo arrays.
// Dword i = { bf16(w[2i]) , bf16(w[2i+1])<<16 } -> exactly the A-fragment
// pair layout of V_WMMA_F32_16X16X32_BF16.
// ---------------------------------------------------------------------------
__global__ void pack_w_kernel(const float* __restrict__ w,
                              uint32_t* __restrict__ whi,
                              uint32_t* __restrict__ wlo, int npairs) {
    int i = blockIdx.x * blockDim.x + threadIdx.x;
    if (i >= npairs) return;
    uint32_t ah, al, bh, bl;
    split_bf16(w[2 * i], ah, al);
    split_bf16(w[2 * i + 1], bh, bl);
    whi[i] = ah | (bh << 16);
    wlo[i] = al | (bl << 16);
}

// ---------------------------------------------------------------------------
// Main CAM GEMM: one workgroup = one (b, spatial-chunk of NP). 8 waves, each
// wave owns a 16-row r-tile (8*16 = R = 128). K-loop in steps of 32 with
// split-bf16 WMMA (hh + hl + lh). Fully software-pipelined:
//  - LDS B-tile double-buffered; next tile's HBM loads issued before the
//    WMMA phase, convert+ds_store after it -> one barrier per K-step.
//  - A fragments (w) double-buffered in registers, prefetched one K-step
//    ahead (K-loop manually unrolled x2 so the ping/pong is register-static).
// Epilogue (LDS cam slab): partial sums/sumsq, optional native cam store,
// optional 2x2 average pooling (chunks are whole row-groups -> WG-local).
// ---------------------------------------------------------------------------
template <int C, int H, int W, int NP, bool POOL, bool STORE_CAM>
__global__ __launch_bounds__(256) void cam_gemm_kernel(
    const float* __restrict__ fmap, const uint32_t* __restrict__ whi,
    const uint32_t* __restrict__ wlo, float* __restrict__ camOut,
    float* __restrict__ poolOut, float* __restrict__ S1out,
    float* __restrict__ S2out, int nchunks) {
    constexpr int  HW    = H * W;
    constexpr int  T     = NP / 16;    // N-tiles per wave
    constexpr int  BSTR  = 20;         // dwords per LDS column (16 pairs + pad)
    constexpr int  BW    = NP * BSTR;  // dwords per hi/lo B array
    constexpr bool VEC4  = (HW % 4 == 0);  // float4 staging possible
    constexpr int  EPG   = VEC4 ? 4 : 1;   // spatial elems per staging group
    constexpr int  G16   = NP / EPG;       // groups per K-pair
    constexpr int  GROUPS = 16 * G16;      // total staging groups
    constexpr int  NIT   = (GROUPS + 255) / 256;
    constexpr int  SMEMB = 4 * BW * 4;     // 2 buffers x (hi+lo)
    constexpr int  SMEMC = 128 * NP * 4;
    constexpr int  SMEM  = SMEMB > SMEMC ? SMEMB : SMEMC;
    static_assert((C / 32) % 2 == 0, "K-loop unrolled x2");
    __shared__ __align__(16) char smem[SMEM];
    uint32_t* Bbase  = (uint32_t*)smem;  // live during K-loop
    float*    camLds = (float*)smem;     // live after K-loop (union)

    const int tid   = threadIdx.x;
    const int wave  = tid >> 5;
    const int lane  = tid & 31;
    const int half  = lane >> 4;  // K-half selector (wave32 WMMA layout)
    const int l16   = lane & 15;
    const int chunk = blockIdx.x;
    const int b     = blockIdx.y;
    const int p0    = chunk * NP;
    const int r0    = wave * 16;

    union Frag { v16bf bf; uint4 u4[2]; };

    v8f acc[T];
#pragma unroll
    for (int t = 0; t < T; ++t)
#pragma unroll
        for (int v = 0; v < 8; ++v) acc[t][v] = 0.f;

    const float* fb = fmap + (size_t)b * C * HW;

    // staged next-B-tile values live in registers between issue and store
    float s0[NIT * EPG], s1[NIT * EPG];

    // issue global loads for B K-tile k (pair rows k+2j, k+2j+1)
    auto issueB = [&](int k) {
#pragma unroll
        for (int it = 0; it < NIT; ++it) {
#pragma unroll
            for (int e = 0; e < EPG; ++e) { s0[it * EPG + e] = 0.f; s1[it * EPG + e] = 0.f; }
            int idx = tid + it * 256;
            if (idx < GROUPS) {
                int j = idx / G16, g = idx - j * G16;
                int p = p0 + EPG * g;
                if (p < HW) {  // VEC4: all-or-nothing (HW%4==0, p%4==0)
                    if constexpr (VEC4) {
                        *(float4*)&s0[it * 4] = *(const float4*)(fb + (size_t)(k + 2 * j) * HW + p);
                        *(float4*)&s1[it * 4] = *(const float4*)(fb + (size_t)(k + 2 * j + 1) * HW + p);
                    } else {
                        s0[it] = fb[(size_t)(k + 2 * j) * HW + p];
                        s1[it] = fb[(size_t)(k + 2 * j + 1) * HW + p];
                    }
                }
            }
        }
    };
    // split-bf16 convert + pack pairs, store into LDS buffer `buf`
    auto storeB = [&](int buf) {
        uint32_t* Bhi = Bbase + buf * 2 * BW;
        uint32_t* Blo = Bhi + BW;
#pragma unroll
        for (int it = 0; it < NIT; ++it) {
            int idx = tid + it * 256;
            if (idx < GROUPS) {
                int j = idx / G16, g = idx - j * G16;
#pragma unroll
                for (int e = 0; e < EPG; ++e) {
                    uint32_t h0, l0, h1, l1;
                    split_bf16(s0[it * EPG + e], h0, l0);
                    split_bf16(s1[it * EPG + e], h1, l1);
                    int col = EPG * g + e;
                    Bhi[col * BSTR + j] = h0 | (h1 << 16);
                    Blo[col * BSTR + j] = l0 | (l1 << 16);
                }
            }
        }
    };
    // A fragments (w rows r0..r0+15, K = k..k+31), a[0]=hi, a[1]=lo
    auto loadA = [&](int k, Frag* a) {
        int base = (r0 + l16) * (C / 2) + (k >> 1) + half * 4;
        const uint4* ph = (const uint4*)(whi + base);
        const uint4* pl = (const uint4*)(wlo + base);
        a[0].u4[0] = ph[0]; a[0].u4[1] = ph[2];  // K 0-7 / 16-23 (lanes<16)
        a[1].u4[0] = pl[0]; a[1].u4[1] = pl[2];  // K 8-15 / 24-31 (lanes>=16)
    };
    // WMMA phase on LDS buffer `buf` with A fragments `a`
    auto compute = [&](int buf, const Frag* a) {
        const uint32_t* Bhi = Bbase + buf * 2 * BW;
        const uint32_t* Blo = Bhi + BW;
#pragma unroll
        for (int t = 0; t < T; ++t) {
            Frag bhi, blo;
            const uint4* qh = (const uint4*)(Bhi + ((t * 16 + l16) * BSTR + half * 4));
            const uint4* ql = (const uint4*)(Blo + ((t * 16 + l16) * BSTR + half * 4));
            bhi.u4[0] = qh[0]; bhi.u4[1] = qh[2];
            blo.u4[0] = ql[0]; blo.u4[1] = ql[2];
            // split-bf16 fp32 emulation: hh + hl + lh
            acc[t] = __builtin_amdgcn_wmma_f32_16x16x32_bf16(
                false, a[0].bf, false, bhi.bf, (short)0, acc[t], false, false);
            acc[t] = __builtin_amdgcn_wmma_f32_16x16x32_bf16(
                false, a[0].bf, false, blo.bf, (short)0, acc[t], false, false);
            acc[t] = __builtin_amdgcn_wmma_f32_16x16x32_bf16(
                false, a[1].bf, false, bhi.bf, (short)0, acc[t], false, false);
        }
    };

    // prologue: stage B K-tile 0 into buffer 0, A fragments for K-tile 0
    Frag aPing[2], aPong[2];
    issueB(0);
    storeB(0);
    loadA(0, aPing);

    for (int k = 0; k < C; k += 64) {
        // ---- sub-iter 0: tile k (LDS buf 0, aPing); prefetch tile k+32
        issueB(k + 32);        // HBM loads in flight across WMMA phase
        loadA(k + 32, aPong);  // L2 loads in flight across WMMA phase
        __syncthreads();       // buf 0 staged by all waves
        compute(0, aPing);
        storeB(1);
        // ---- sub-iter 1: tile k+32 (LDS buf 1, aPong); prefetch tile k+64
        const bool more = (k + 64 < C);
        if (more) {
            issueB(k + 64);
            loadA(k + 64, aPing);
        }
        __syncthreads();       // buf 1 staged by all waves
        compute(1, aPong);
        if (more) storeB(0);
    }

    __syncthreads();  // retire B buffers, repurpose LDS as cam slab
    // D layout: lanes 0-15 -> M=v, lanes 16-31 -> M=v+8 ; N = lane%16
#pragma unroll
    for (int t = 0; t < T; ++t)
#pragma unroll
        for (int v = 0; v < 8; ++v)
            camLds[(r0 + v + 8 * half) * NP + t * 16 + l16] = acc[t][v];
    __syncthreads();

    // ---- partial sums for embedding / certainty
    if (tid < 128) {
        float ps1 = 0.f, ps2 = 0.f;
        for (int pp = 0; pp < NP; ++pp) {
            if (p0 + pp < HW) {
                float v = camLds[tid * NP + pp];
                ps1 += v;
                ps2 += v * v;
            }
        }
        size_t o = ((size_t)b * 128 + tid) * nchunks + chunk;
        S1out[o] = ps1;
        S2out[o] = ps2;
    }
    // ---- native-resolution cam (link "hi" side)
    if constexpr (STORE_CAM) {
        for (int idx = tid; idx < 128 * NP; idx += 256) {
            int r = idx / NP, pp = idx - r * NP;
            int p = p0 + pp;
            if (p < HW)
                camOut[((size_t)b * 128 + r) * HW + p] = camLds[r * NP + pp];
        }
    }
    // ---- 2x2 average pool (link "lo" side)
    if constexpr (POOL) {
        constexpr int ROWS  = NP / W;  // input rows in this chunk (even)
        constexpr int PROWS = ROWS / 2;
        constexpr int PW    = W / 2;
        const int ph0 = chunk * PROWS;
        for (int idx = tid; idx < 128 * PROWS * PW; idx += 256) {
            int r   = idx / (PROWS * PW);
            int rem = idx - r * (PROWS * PW);
            int pi  = rem / PW;
            int pj  = rem - pi * PW;
            int ph  = ph0 + pi;
            if (ph < H / 2) {
                int base = (2 * pi) * W + 2 * pj;
                float v = 0.25f * (camLds[r * NP + base] + camLds[r * NP + base + 1] +
                                   camLds[r * NP + base + W] + camLds[r * NP + base + W + 1]);
                poolOut[((size_t)b * 128 + r) * ((H / 2) * (W / 2)) + ph * PW + pj] = v;
            }
        }
    }
}

// ---------------------------------------------------------------------------
// Reduce partial sums -> embedding (= sum/HW + bias) and certainty (ddof=1 std)
// ---------------------------------------------------------------------------
__global__ void finalize_kernel(const float* __restrict__ S1p,
                                const float* __restrict__ S2p,
                                const float* __restrict__ bias, int nchunks,
                                int HW, float* __restrict__ emb,
                                float* __restrict__ cert) {
    int i = blockIdx.x * blockDim.x + threadIdx.x;  // i = b*128 + r
    if (i >= 64 * 128) return;
    float s1 = 0.f, s2 = 0.f;
    for (int c = 0; c < nchunks; ++c) {
        s1 += S1p[(size_t)i * nchunks + c];
        s2 += S2p[(size_t)i * nchunks + c];
    }
    float n  = (float)HW;
    emb[i]   = s1 / n + bias[i & 127];
    float vr = (s2 - s1 * s1 / n) / (n - 1.f);
    cert[i]  = sqrtf(fmaxf(vr, 0.f));
}

// ---------------------------------------------------------------------------
// Per-(b,r) L2-normalize over spatial and transpose to [r][b*S + i]
// (K-major layout for the link GEMM). One wave32 per (b,r).
// ---------------------------------------------------------------------------
__global__ void normalize_T_kernel(const float* __restrict__ src,
                                   float* __restrict__ dst, int S) {
    int pair = blockIdx.x * 8 + (threadIdx.x >> 5);  // pair = b*128 + r
    int lane = threadIdx.x & 31;
    int bb   = pair >> 7;
    int r    = pair & 127;
    const float* s = src + (size_t)pair * S;
    float ss = 0.f;
    for (int i = lane; i < S; i += 32) { float v = s[i]; ss += v * v; }
#pragma unroll
    for (int m = 16; m > 0; m >>= 1) ss += __shfl_xor(ss, m, 32);
    float scale = 1.f / fmaxf(sqrtf(ss), 1e-12f);
    float* d = dst + (size_t)r * (64 * (size_t)S) + (size_t)bb * S;
    for (int i = lane; i < S; i += 32) d[i] = s[i] * scale;
}

// ---------------------------------------------------------------------------
// link[m,n] = (1/B) * sum_{b,i} lo[m][b,i] * hi[n][b,i]  via fp32 WMMA
// (tiny, L2-resident: K = 64*S). grid.x = m-tile, wave = n-tile.
// 4 interleaved accumulators break the serial WMMA dependency chain.
// ---------------------------------------------------------------------------
template <int S>
__global__ __launch_bounds__(256) void link_kernel(
    const float* __restrict__ loT, const float* __restrict__ hiT,
    float* __restrict__ out) {
    constexpr int K = 64 * S;  // 12544 or 3136, both % 16 == 0
    const int lane = threadIdx.x & 31;
    const int half = lane >> 4;
    const int l16  = lane & 15;
    const int m0   = blockIdx.x * 16;
    const int n0   = (threadIdx.x >> 5) * 16;
    const float* arow = loT + (size_t)(m0 + l16) * K + 2 * half;
    const float* brow = hiT + (size_t)(n0 + l16) * K + 2 * half;
    v8f acc[4];
#pragma unroll
    for (int u = 0; u < 4; ++u)
#pragma unroll
        for (int v = 0; v < 8; ++v) acc[u][v] = 0.f;
    for (int k = 0; k < K; k += 16) {
#pragma unroll
        for (int u = 0; u < 4; ++u) {
            v2f a  = *(const v2f*)(arow + k + 4 * u);
            v2f bv = *(const v2f*)(brow + k + 4 * u);
            acc[u] = __builtin_amdgcn_wmma_f32_16x16x4_f32(
                false, a, false, bv, (short)0, acc[u], false, false);
        }
    }
#pragma unroll
    for (int v = 0; v < 8; ++v) {
        int m = m0 + v + 8 * half;
        out[m * 128 + n0 + l16] =
            (acc[0][v] + acc[1][v] + acc[2][v] + acc[3][v]) * (1.f / 64.f);
    }
}

// ---------------------------------------------------------------------------
extern "C" void kernel_launch(void* const* d_in, const int* in_sizes, int n_in,
                              void* d_out, int out_size, void* d_ws,
                              size_t ws_size, hipStream_t stream) {
    const float* fmap0 = (const float*)d_in[0];
    const float* w0    = (const float*)d_in[1];
    const float* b0    = (const float*)d_in[2];
    const float* fmap1 = (const float*)d_in[3];
    const float* w1    = (const float*)d_in[4];
    const float* b1    = (const float*)d_in[5];
    const float* fmap2 = (const float*)d_in[6];
    const float* w2    = (const float*)d_in[7];
    const float* b2    = (const float*)d_in[8];
    float* out = (float*)d_out;

    char*  ws  = (char*)d_ws;
    size_t off = 0;
    auto alloc = [&](size_t bytes) -> void* {
        void* p = ws + off;
        off = (off + bytes + 255) & ~(size_t)255;
        return p;
    };
    uint32_t* whi0 = (uint32_t*)alloc(128 * 256 * 4);
    uint32_t* wlo0 = (uint32_t*)alloc(128 * 256 * 4);
    uint32_t* whi1 = (uint32_t*)alloc(128 * 512 * 4);
    uint32_t* wlo1 = (uint32_t*)alloc(128 * 512 * 4);
    uint32_t* whi2 = (uint32_t*)alloc(128 * 1024 * 4);
    uint32_t* wlo2 = (uint32_t*)alloc(128 * 1024 * 4);
    float* cam1  = (float*)alloc((size_t)64 * 128 * 196 * 4);
    float* cam2  = (float*)alloc((size_t)64 * 128 * 49 * 4);
    float* pool0 = (float*)alloc((size_t)64 * 128 * 196 * 4);
    float* pool1 = (float*)alloc((size_t)64 * 128 * 49 * 4);
    float* S1p0  = (float*)alloc((size_t)64 * 128 * 7 * 4);
    float* S2p0  = (float*)alloc((size_t)64 * 128 * 7 * 4);
    float* S1p1  = (float*)alloc((size_t)64 * 128 * 2 * 4);
    float* S2p1  = (float*)alloc((size_t)64 * 128 * 2 * 4);
    float* S1p2  = (float*)alloc((size_t)64 * 128 * 4);
    float* S2p2  = (float*)alloc((size_t)64 * 128 * 4);
    float* lo0T  = (float*)alloc((size_t)128 * 12544 * 4);
    float* hi1T  = (float*)alloc((size_t)128 * 12544 * 4);
    float* lo1T  = (float*)alloc((size_t)128 * 3136 * 4);
    float* hi2T  = (float*)alloc((size_t)128 * 3136 * 4);

    // 1) pack w -> split-bf16 A-fragment layout
    pack_w_kernel<<<(128 * 256 + 255) / 256, 256, 0, stream>>>(w0, whi0, wlo0, 128 * 256);
    pack_w_kernel<<<(128 * 512 + 255) / 256, 256, 0, stream>>>(w1, whi1, wlo1, 128 * 512);
    pack_w_kernel<<<(128 * 1024 + 255) / 256, 256, 0, stream>>>(w2, whi2, wlo2, 128 * 1024);

    // 2) CAM GEMMs (bf16-split WMMA) with fused sums/pool epilogues
    cam_gemm_kernel<512, 28, 28, 112, true, false>
        <<<dim3(7, 64), 256, 0, stream>>>(fmap0, whi0, wlo0, nullptr, pool0, S1p0, S2p0, 7);
    cam_gemm_kernel<1024, 14, 14, 112, true, true>
        <<<dim3(2, 64), 256, 0, stream>>>(fmap1, whi1, wlo1, cam1, pool1, S1p1, S2p1, 2);
    cam_gemm_kernel<2048, 7, 7, 64, false, true>
        <<<dim3(1, 64), 256, 0, stream>>>(fmap2, whi2, wlo2, cam2, nullptr, S1p2, S2p2, 1);

    // 3) embeddings + certainties
    finalize_kernel<<<32, 256, 0, stream>>>(S1p0, S2p0, b0, 7, 784, out + 0,     out + 24576);
    finalize_kernel<<<32, 256, 0, stream>>>(S1p1, S2p1, b1, 2, 196, out + 8192,  out + 32768);
    finalize_kernel<<<32, 256, 0, stream>>>(S1p2, S2p2, b2, 1, 49,  out + 16384, out + 40960);

    // 4) normalize + transpose for links
    normalize_T_kernel<<<1024, 256, 0, stream>>>(pool0, lo0T, 196);
    normalize_T_kernel<<<1024, 256, 0, stream>>>(cam1,  hi1T, 196);
    normalize_T_kernel<<<1024, 256, 0, stream>>>(pool1, lo1T, 49);
    normalize_T_kernel<<<1024, 256, 0, stream>>>(cam2,  hi2T, 49);

    // 5) links via fp32 WMMA
    link_kernel<196><<<8, 256, 0, stream>>>(lo0T, hi1T, out + 49152);
    link_kernel<49> <<<8, 256, 0, stream>>>(lo1T, hi2T, out + 65536);
}
